// SwinTransformerBlock_67637144977909
// MI455X (gfx1250) — matrix operand order repaired and checked by reading, not compiled, not run on hardware
//
#include <hip/hip_runtime.h>
#include <math.h>

// ---------------- CDNA5 WMMA types / helpers ----------------
typedef __attribute__((ext_vector_type(16))) _Float16 v16h;
typedef __attribute__((ext_vector_type(8)))  float    v8f;

__device__ __forceinline__ v8f wmma32(v16h a, v16h b, v8f c) {
  return __builtin_amdgcn_wmma_f32_16x16x32_f16(
      false, a, false, b, (short)0, c, false, false);
}

// A fragment: rows [row0,row0+16), K window [k0,k0+32), from row-major [M][ld] f16.
__device__ __forceinline__ v16h load_A(const _Float16* p, int row0, int ld, int k0, int lane) {
  const _Float16* q = p + (row0 + (lane & 15)) * ld + k0 + ((lane >> 4) ? 8 : 0);
  v16h a;
  *(uint4*)&a       = *(const uint4*)q;
  *((uint4*)&a + 1) = *(const uint4*)(q + 16);
  return a;
}

// B fragment (32x16 KxN) from buffer indexed [n][k] (row-major [N][ld]).
__device__ __forceinline__ v16h load_Bt(const _Float16* p, int col0, int ld, int k0, int lane) {
  const _Float16* q = p + (col0 + (lane & 15)) * ld + k0 + ((lane >> 4) ? 8 : 0);
  v16h b;
  *(uint4*)&b       = *(const uint4*)q;
  *((uint4*)&b + 1) = *(const uint4*)(q + 16);
  return b;
}

union pack8 { uint4 u; _Float16 e[8]; };

// ---- CDNA5 async global->LDS copy (ASYNCcnt-tracked, overlaps with compute) ----
__device__ __forceinline__ void async_copy_lds(_Float16* dst, const _Float16* src,
                                               int n_f16, int tid, int nthr) {
  int nv = n_f16 >> 3;  // 16B units
  for (int i = tid; i < nv; i += nthr) {
    unsigned loff = (unsigned)(uintptr_t)(dst + i * 8);   // LDS byte address = low 32 bits
    const void* g = (const void*)(src + i * 8);
    asm volatile("global_load_async_to_lds_b128 %0, %1, off"
                 :: "v"(loff), "v"(g) : "memory");
  }
}
__device__ __forceinline__ void wait_async() {
  asm volatile("s_wait_asynccnt 0x0" ::: "memory");
}

__device__ __forceinline__ int regidx(int r) { return (r < 120) ? 0 : ((r < 124) ? 1 : 2); }

#define SHIFT_ 4

// ---------------- kernel 0: f32 -> f16 transpose+pad weight conversion ----------------
__global__ void k_cvtT(const float* __restrict__ src, _Float16* __restrict__ dst,
                       int K, int N, int ldp) {
  int i = blockIdx.x * 256 + threadIdx.x;
  if (i < K * N) {
    int k = i / N, n = i - k * N;
    dst[n * ldp + k] = (_Float16)src[i];
  }
}

// ---------------- kernel 1: fused LN1 + shift/partition + window attention + proj + residual ----
// grid 2048 windows, block 128 (4 waves; wave w owns token tile tw=16w)
// LDS (f16): xw 64x104 | WqkvT 288x104 | WprojT 96x104 | Q 64x40 | K 64x40
//            Vt 32x72 | P 64x72 | O 64x104  => 65280 f16 = 130560 B
__global__ void k_attn(const float* __restrict__ x_in, const float* __restrict__ g1,
                       const float* __restrict__ b1,
                       const _Float16* __restrict__ wqkvT_g, const _Float16* __restrict__ wprojT_g,
                       const float* __restrict__ bqkv, const float* __restrict__ bproj,
                       const float* __restrict__ bias_table, float* __restrict__ x1) {
  extern __shared__ _Float16 sm[];
  _Float16* xw = sm;                 // 6656
  _Float16* wq = xw + 6656;          // 29952
  _Float16* wp = wq + 29952;         // 9984
  _Float16* qs = wp + 9984;          // 2560
  _Float16* ks = qs + 2560;          // 2560
  _Float16* vt = ks + 2560;          // 2304
  _Float16* ps = vt + 2304;          // 4608
  _Float16* os = ps + 4608;          // 6656

  int widx = blockIdx.x;
  int b = widx >> 8, wh = (widx >> 4) & 15, ww = widx & 15;
  int tid = threadIdx.x, lane = tid & 31, wv = tid >> 5;
  int n = lane & 15, h = lane >> 4, tw = wv * 16;
  const float scale = 0.17677669529663687f;  // 1/sqrt(32)

  // kick off async weight staging; it overlaps with the LN phase below
  async_copy_lds(wq, wqkvT_g, 29952, tid, 128);
  async_copy_lds(wp, wprojT_g, 9984, tid, 128);

  // LN1 on this wave's 16 tokens (shifted source coords) -> xw row-major f16
  for (int t = tw; t < tw + 16; ++t) {
    int i = t >> 3, j = t & 7;
    int r = (wh * 8 + i + SHIFT_) & 127;
    int c = (ww * 8 + j + SHIFT_) & 127;
    const float* src = x_in + ((size_t)b * 16384 + r * 128 + c) * 96;
    float v0 = src[lane], v1 = src[lane + 32], v2 = src[lane + 64];
    float s = v0 + v1 + v2, s2 = v0 * v0 + v1 * v1 + v2 * v2;
#pragma unroll
    for (int o = 16; o >= 1; o >>= 1) {
      s += __shfl_xor(s, o, 32);
      s2 += __shfl_xor(s2, o, 32);
    }
    float mu = s * (1.0f / 96.0f);
    float var = s2 * (1.0f / 96.0f) - mu * mu;
    float rs = rsqrtf(var + 1e-5f);
    xw[t * 104 + lane]      = (_Float16)((v0 - mu) * rs * g1[lane]      + b1[lane]);
    xw[t * 104 + lane + 32] = (_Float16)((v1 - mu) * rs * g1[lane + 32] + b1[lane + 32]);
    xw[t * 104 + lane + 64] = (_Float16)((v2 - mu) * rs * g1[lane + 64] + b1[lane + 64]);
  }
  wait_async();
  __syncthreads();

  // hoisted fragments of the LN'd window input
  v16h axw[3], bxw[3];
#pragma unroll
  for (int kk = 0; kk < 3; ++kk) {
    axw[kk] = load_A(xw, tw, 104, kk * 32, lane);
    bxw[kk] = load_Bt(xw, tw, 104, kk * 32, lane);
  }

  int myq = tw + n;
  int qi = myq >> 3, qj = myq & 7;
  int lq = regidx(wh * 8 + qi) * 3 + regidx(ww * 8 + qj);

  for (int hd = 0; hd < 3; ++hd) {
    // ---- Q^T, K^T = Wqkv^T @ xw^T : packed stores into row-major [token][40] ----
#pragma unroll
    for (int p = 0; p < 2; ++p) {
      _Float16* dst = p ? ks : qs;
      int cb = p * 96 + hd * 32;
#pragma unroll
      for (int mt = 0; mt < 2; ++mt) {
        v8f acc = {};
#pragma unroll
        for (int kk = 0; kk < 3; ++kk)
          acc = wmma32(load_A(wq, cb + mt * 16, 104, kk * 32, lane), bxw[kk], acc);
        const float* bp_ = bqkv + cb + mt * 16 + 8 * h;
        float4 b0 = *(const float4*)bp_, b1v = *(const float4*)(bp_ + 4);
        float bb[8] = {b0.x, b0.y, b0.z, b0.w, b1v.x, b1v.y, b1v.z, b1v.w};
        pack8 pk;
#pragma unroll
        for (int v = 0; v < 8; ++v) {
          float val = acc[v] + bb[v];
          if (p == 0) val *= scale;  // fold softmax scale into Q
          pk.e[v] = (_Float16)val;
        }
        *(uint4*)(dst + myq * 40 + mt * 16 + 8 * h) = pk.u;
      }
    }
    // ---- V = xw @ Wv : packed stores into transposed [chan][72] ----
    {
      int cb = 192 + hd * 32;
#pragma unroll
      for (int nt = 0; nt < 2; ++nt) {
        v8f acc = {};
#pragma unroll
        for (int kk = 0; kk < 3; ++kk)
          acc = wmma32(axw[kk], load_Bt(wq, cb + nt * 16, 104, kk * 32, lane), acc);
        float bias = bqkv[cb + nt * 16 + n];
        pack8 pk;
#pragma unroll
        for (int v = 0; v < 8; ++v) pk.e[v] = (_Float16)(acc[v] + bias);
        *(uint4*)(vt + (nt * 16 + n) * 72 + tw + 8 * h) = pk.u;
      }
    }
    __syncthreads();

    // ---- S^T = K @ Q^T : cols = q tokens (lane-fixed), rows = k tokens ----
    v16h bq = load_Bt(qs, tw, 40, 0, lane);
    v8f sf[4];
#pragma unroll
    for (int kmt = 0; kmt < 4; ++kmt) {
      v8f z = {};
      sf[kmt] = wmma32(load_A(ks, kmt * 16, 40, 0, lane), bq, z);
    }
#pragma unroll
    for (int kmt = 0; kmt < 4; ++kmt) {
#pragma unroll
      for (int v = 0; v < 8; ++v) {
        int tk = kmt * 16 + v + 8 * h;
        int ki = tk >> 3, kj = tk & 7;
        float bias = bias_table[((qi - ki + 7) * 15 + (qj - kj + 7)) * 3 + hd];
        int lk = regidx(wh * 8 + ki) * 3 + regidx(ww * 8 + kj);
        sf[kmt][v] += bias + ((lq != lk) ? -100.0f : 0.0f);
      }
    }
    // softmax over k: 32 in-lane values + partner lane (same q, other 32 k's)
    float mx = -1e30f;
#pragma unroll
    for (int kmt = 0; kmt < 4; ++kmt)
#pragma unroll
      for (int v = 0; v < 8; ++v) mx = fmaxf(mx, sf[kmt][v]);
    mx = fmaxf(mx, __shfl_xor(mx, 16, 32));
    float sum = 0.0f;
#pragma unroll
    for (int kmt = 0; kmt < 4; ++kmt)
#pragma unroll
      for (int v = 0; v < 8; ++v) {
        sf[kmt][v] = __expf(sf[kmt][v] - mx);
        sum += sf[kmt][v];
      }
    sum += __shfl_xor(sum, 16, 32);
    float inv = 1.0f / sum;
#pragma unroll
    for (int kmt = 0; kmt < 4; ++kmt) {
      pack8 pk;
#pragma unroll
      for (int v = 0; v < 8; ++v) pk.e[v] = (_Float16)(sf[kmt][v] * inv);
      *(uint4*)(ps + myq * 72 + kmt * 16 + 8 * h) = pk.u;
    }

    // ---- O^T = V^T @ P^T : packed store into os row-major [token][104] ----
    v16h bp0 = load_Bt(ps, tw, 72, 0, lane);
    v16h bp1 = load_Bt(ps, tw, 72, 32, lane);
#pragma unroll
    for (int mt = 0; mt < 2; ++mt) {
      v8f acc = {};
      acc = wmma32(load_A(vt, mt * 16, 72, 0, lane), bp0, acc);
      acc = wmma32(load_A(vt, mt * 16, 72, 32, lane), bp1, acc);
      pack8 pk;
#pragma unroll
      for (int v = 0; v < 8; ++v) pk.e[v] = (_Float16)acc[v];
      *(uint4*)(os + myq * 104 + hd * 32 + mt * 16 + 8 * h) = pk.u;
    }
    __syncthreads();
  }

  // ---- proj^T = Wproj^T @ O^T + residual : vectorized global I/O ----
  v16h bos[3];
#pragma unroll
  for (int kk = 0; kk < 3; ++kk) bos[kk] = load_Bt(os, tw, 104, kk * 32, lane);
  int r = (wh * 8 + qi + SHIFT_) & 127;
  int c = (ww * 8 + qj + SHIFT_) & 127;
  const float* srcp = x_in + ((size_t)b * 16384 + r * 128 + c) * 96;
  float* dstp = x1 + ((size_t)b * 16384 + r * 128 + c) * 96;
#pragma unroll
  for (int mt = 0; mt < 6; ++mt) {
    v8f acc = {};
#pragma unroll
    for (int kk = 0; kk < 3; ++kk)
      acc = wmma32(load_A(wp, mt * 16, 104, kk * 32, lane), bos[kk], acc);
    int ch = mt * 16 + 8 * h;
    float4 bb0 = *(const float4*)(bproj + ch), bb1 = *(const float4*)(bproj + ch + 4);
    float4 x0 = *(const float4*)(srcp + ch), x1v = *(const float4*)(srcp + ch + 4);
    float4 o0, o1;
    o0.x = x0.x + acc[0] + bb0.x; o0.y = x0.y + acc[1] + bb0.y;
    o0.z = x0.z + acc[2] + bb0.z; o0.w = x0.w + acc[3] + bb0.w;
    o1.x = x1v.x + acc[4] + bb1.x; o1.y = x1v.y + acc[5] + bb1.y;
    o1.z = x1v.z + acc[6] + bb1.z; o1.w = x1v.w + acc[7] + bb1.w;
    *(float4*)(dstp + ch) = o0;
    *(float4*)(dstp + ch + 4) = o1;
  }
}

// ---------------- kernel 2: fused LN2 + MLP + residual, in-place on d_out ----------------
// grid 2048 (64-row blocks), block 128
// LDS: h2 64x104 f16 (13312B) | act 64x392 f16 (50176B) | wbuf 39936 f16 (79872B) = 143360 B
__global__ void k_mlp(const float* __restrict__ g2, const float* __restrict__ b2,
                      const _Float16* __restrict__ w1T_g, const float* __restrict__ bm1,
                      const _Float16* __restrict__ w2T_g, const float* __restrict__ bm2,
                      float* __restrict__ io) {
  extern __shared__ char smraw[];
  _Float16* h2   = (_Float16*)smraw;                    // 6656 f16
  _Float16* act  = (_Float16*)(smraw + 13312);          // 25088 f16
  _Float16* wbuf = (_Float16*)(smraw + 13312 + 50176);  // 39936 f16

  int tid = threadIdx.x, lane = tid & 31, wv = tid >> 5;
  int n = lane & 15, h = lane >> 4, tw = wv * 16;
  size_t r0 = (size_t)blockIdx.x * 64;

  // async-stage W1 while LN2 runs
  async_copy_lds(wbuf, w1T_g, 39936, tid, 128);

  // LN2 straight from global x1 (= io)
  for (int t = tw; t < tw + 16; ++t) {
    const float* src = io + (r0 + t) * 96;
    float v0 = src[lane], v1 = src[lane + 32], v2 = src[lane + 64];
    float s = v0 + v1 + v2, s2 = v0 * v0 + v1 * v1 + v2 * v2;
#pragma unroll
    for (int o = 16; o >= 1; o >>= 1) {
      s += __shfl_xor(s, o, 32);
      s2 += __shfl_xor(s2, o, 32);
    }
    float mu = s * (1.0f / 96.0f);
    float var = s2 * (1.0f / 96.0f) - mu * mu;
    float rs = rsqrtf(var + 1e-5f);
    h2[t * 104 + lane]      = (_Float16)((v0 - mu) * rs * g2[lane]      + b2[lane]);
    h2[t * 104 + lane + 32] = (_Float16)((v1 - mu) * rs * g2[lane + 32] + b2[lane + 32]);
    h2[t * 104 + lane + 64] = (_Float16)((v2 - mu) * rs * g2[lane + 64] + b2[lane + 64]);
  }
  wait_async();
  __syncthreads();

  int mytok = tw + n;

  // GEMM1^T: act^T = W1^T @ h2^T, + bm1, exact GELU; packed store -> act row-major [token][392]
  v16h bh[3];
#pragma unroll
  for (int kk = 0; kk < 3; ++kk) bh[kk] = load_Bt(h2, tw, 104, kk * 32, lane);
  for (int mt = 0; mt < 24; ++mt) {
    v8f acc = {};
#pragma unroll
    for (int kk = 0; kk < 3; ++kk)
      acc = wmma32(load_A(wbuf, mt * 16, 104, kk * 32, lane), bh[kk], acc);
    int hb_ = mt * 16 + 8 * h;
    float4 b0 = *(const float4*)(bm1 + hb_), b1v = *(const float4*)(bm1 + hb_ + 4);
    float bb[8] = {b0.x, b0.y, b0.z, b0.w, b1v.x, b1v.y, b1v.z, b1v.w};
    pack8 pk;
#pragma unroll
    for (int v = 0; v < 8; ++v) {
      float u = acc[v] + bb[v];
      pk.e[v] = (_Float16)(0.5f * u * (1.0f + erff(u * 0.70710678118654752f)));
    }
    *(uint4*)(act + mytok * 392 + hb_) = pk.u;
  }
  __syncthreads();
  async_copy_lds(wbuf, w2T_g, 37632, tid, 128);
  wait_async();
  __syncthreads();

  // GEMM2^T: out^T = W2^T @ act^T + bm2 + x1 : vectorized in-place global update
  v16h ba[12];
#pragma unroll
  for (int kk = 0; kk < 12; ++kk) ba[kk] = load_Bt(act, tw, 392, kk * 32, lane);
  float* iop = io + (r0 + mytok) * 96;
#pragma unroll
  for (int mt = 0; mt < 6; ++mt) {
    v8f acc = {};
#pragma unroll
    for (int kk = 0; kk < 12; ++kk)
      acc = wmma32(load_A(wbuf, mt * 16, 392, kk * 32, lane), ba[kk], acc);
    int ch = mt * 16 + 8 * h;
    float4 b0 = *(const float4*)(bm2 + ch), b1v = *(const float4*)(bm2 + ch + 4);
    float4 x0 = *(const float4*)(iop + ch), x1v = *(const float4*)(iop + ch + 4);
    float4 o0, o1;
    o0.x = x0.x + acc[0] + b0.x; o0.y = x0.y + acc[1] + b0.y;
    o0.z = x0.z + acc[2] + b0.z; o0.w = x0.w + acc[3] + b0.w;
    o1.x = x1v.x + acc[4] + b1v.x; o1.y = x1v.y + acc[5] + b1v.y;
    o1.z = x1v.z + acc[6] + b1v.z; o1.w = x1v.w + acc[7] + b1v.w;
    *(float4*)(iop + ch) = o0;
    *(float4*)(iop + ch + 4) = o1;
  }
}

// ---------------- launcher ----------------
extern "C" void kernel_launch(void* const* d_in, const int* in_sizes, int n_in,
                              void* d_out, int out_size, void* d_ws, size_t ws_size,
                              hipStream_t stream) {
  (void)in_sizes; (void)n_in; (void)out_size; (void)ws_size;
  const float* x     = (const float*)d_in[0];
  const float* g1    = (const float*)d_in[1];
  const float* b1    = (const float*)d_in[2];
  const float* Wqkv  = (const float*)d_in[3];
  const float* bqkv  = (const float*)d_in[4];
  const float* Wproj = (const float*)d_in[5];
  const float* bproj = (const float*)d_in[6];
  const float* btab  = (const float*)d_in[7];
  const float* g2    = (const float*)d_in[8];
  const float* b2    = (const float*)d_in[9];
  const float* W1    = (const float*)d_in[10];
  const float* bm1   = (const float*)d_in[11];
  const float* W2    = (const float*)d_in[12];
  const float* bm2   = (const float*)d_in[13];
  float* out = (float*)d_out;

  char* ws = (char*)d_ws;
  size_t off = 0;
  _Float16* wqkvT  = (_Float16*)(ws + off);  off += (size_t)288 * 104 * 2;  // 59904
  _Float16* wprojT = (_Float16*)(ws + off);  off += (size_t)96 * 104 * 2;   // 19968
  _Float16* w1T    = (_Float16*)(ws + off);  off += (size_t)384 * 104 * 2;  // 79872
  _Float16* w2T    = (_Float16*)(ws + off);  off += (size_t)96 * 392 * 2;   // 75264

  k_cvtT<<<(96 * 288 + 255) / 256, 256, 0, stream>>>(Wqkv, wqkvT, 96, 288, 104);
  k_cvtT<<<(96 * 96 + 255) / 256, 256, 0, stream>>>(Wproj, wprojT, 96, 96, 104);
  k_cvtT<<<(96 * 384 + 255) / 256, 256, 0, stream>>>(W1, w1T, 96, 384, 104);
  k_cvtT<<<(384 * 96 + 255) / 256, 256, 0, stream>>>(W2, w2T, 384, 96, 392);

  k_attn<<<2048, 128, 130560, stream>>>(x, g1, b1, wqkvT, wprojT, bqkv, bproj, btab, out);
  k_mlp<<<2048, 128, 143360, stream>>>(g2, b2, w1T, bm1, w2T, bm2, out);
}
